// POI2Region_1142461301192
// MI455X (gfx1250) — matrix operand order repaired and checked by reading, not compiled, not run on hardware
//
#include <hip/hip_runtime.h>
#include <math.h>

typedef __attribute__((ext_vector_type(16))) __bf16 v16bf;
typedef __attribute__((ext_vector_type(8)))  float  v8f;

#define RSQRT128 0.08838834764831845f

__device__ __forceinline__ void atomAddF(float* p, float v) {
  unsafeAtomicAdd(p, v);   // lowers to global_atomic_add_f32 on gfx1250
}

// Build a 16x32 bf16 A-fragment for one lane from a row-major f32 row.
// Lane layout (ISA 7.12.2): row = lane&15, kbase = (lane>>4)*8,
// elements 0..7 -> K = kt*32 + kbase + e ; 8..15 -> K = kt*32 + 16 + kbase + (e-8).
__device__ __forceinline__ v16bf make_afrag(const float* __restrict__ xr,
                                            int kt, int kbase) {
  const float* p0 = xr + kt * 32 + kbase;
  float4 c0 = *(const float4*)p0;
  float4 c1 = *(const float4*)(p0 + 4);
  float4 c2 = *(const float4*)(p0 + 16);
  float4 c3 = *(const float4*)(p0 + 20);
  v16bf a;
  a[0]=(__bf16)c0.x; a[1]=(__bf16)c0.y; a[2]=(__bf16)c0.z; a[3]=(__bf16)c0.w;
  a[4]=(__bf16)c1.x; a[5]=(__bf16)c1.y; a[6]=(__bf16)c1.z; a[7]=(__bf16)c1.w;
  a[8]=(__bf16)c2.x; a[9]=(__bf16)c2.y; a[10]=(__bf16)c2.z; a[11]=(__bf16)c2.w;
  a[12]=(__bf16)c3.x; a[13]=(__bf16)c3.y; a[14]=(__bf16)c3.z; a[15]=(__bf16)c3.w;
  return a;
}

// ---------------------------------------------------------------------------
// Prep: q = S@Wq + bq ; wqv[j][h] = Wk[j, h*32:(h+1)*32] . q_h ; c[h] = bk_h.q_h
// ---------------------------------------------------------------------------
__global__ void k_prep(const float* __restrict__ S, const float* __restrict__ Wq,
                       const float* __restrict__ bq, const float* __restrict__ Wk,
                       const float* __restrict__ bk,
                       float* qv, float* wqv, float* cvec) {
  __shared__ float qs[128];
  int j = threadIdx.x;
  float acc = bq[j];
  for (int k2 = 0; k2 < 128; k2++) acc += S[k2] * Wq[k2 * 128 + j];
  qs[j] = acc;
  qv[j] = acc;
  __syncthreads();
  float w0 = 0.f, w1 = 0.f, w2 = 0.f, w3 = 0.f;
  for (int i = 0; i < 32; i++) {
    w0 += Wk[j * 128 + i]      * qs[i];
    w1 += Wk[j * 128 + 32 + i] * qs[32 + i];
    w2 += Wk[j * 128 + 64 + i] * qs[64 + i];
    w3 += Wk[j * 128 + 96 + i] * qs[96 + i];
  }
  wqv[j * 4 + 0] = w0; wqv[j * 4 + 1] = w1;
  wqv[j * 4 + 2] = w2; wqv[j * 4 + 3] = w3;
  if (j < 4) {
    float c = 0.f;
    for (int i = 0; i < 32; i++) c += bk[j * 32 + i] * qs[j * 32 + i];
    cvec[j] = c;
  }
}

// ---------------------------------------------------------------------------
// Pack Wv/Wo/Wg (f32 row-major 128x128) into per-lane-contiguous bf16 WMMA
// B-fragments: frag(mat,kt,nt): lane holds col = nt*16+(lane&15),
// K = kt*32 + (lane>>4)*8 + e for e<8, kt*32 + 16 + (lane>>4)*8 + (e-8) for e>=8 (one 32B load per lane).
// ---------------------------------------------------------------------------
__global__ void k_pack(const float* __restrict__ Wv, const float* __restrict__ Wo,
                       const float* __restrict__ Wg, __bf16* packed) {
  int t = blockIdx.x * blockDim.x + threadIdx.x;
  if (t >= 3072) return;
  int lane = t & 31;
  int nt   = (t >> 5) & 7;
  int kt   = (t >> 8) & 3;
  int mat  = t >> 10;
  const float* W = (mat == 0) ? Wv : ((mat == 1) ? Wo : Wg);
  int kb  = kt * 32 + ((lane >> 4) * 8);
  int col = nt * 16 + (lane & 15);
  __bf16* dst = packed + (size_t)t * 16;
  for (int e = 0; e < 16; e++) dst[e] = (__bf16)W[(size_t)(kb + e + ((e >> 3) << 3)) * 128 + col];
}

// ---------------------------------------------------------------------------
// Pass A: scores[n,h] = (x[n]·wqv[:,h] + c[h]) * rsqrt(128); atomic seg-max
// via order-preserving uint key.
// ---------------------------------------------------------------------------
__global__ __launch_bounds__(256) void k_scores(
    const float* __restrict__ x, const int* __restrict__ zone,
    const float* __restrict__ wqv, const float* __restrict__ cvec,
    float* __restrict__ scores, unsigned* __restrict__ smaxKey, int N) {
  __shared__ float xs[64 * 128];
  __shared__ float wqs[512];
  int t = threadIdx.x;
  for (int i = t; i < 512; i += 256) wqs[i] = wqv[i];
  size_t base  = (size_t)blockIdx.x * 8192;
  size_t total = (size_t)N * 128;
  for (int i = 0; i < 8; i++) {
    size_t vi = (size_t)(t + i * 256) * 4;
    float4 val = make_float4(0.f, 0.f, 0.f, 0.f);
    if (base + vi + 3 < total) val = *(const float4*)(x + base + vi);
    *(float4*)&xs[vi] = val;
  }
  __syncthreads();
  int row = t >> 2, h = t & 3;
  int n = blockIdx.x * 64 + row;
  if (n < N) {
    float acc = 0.f;
    const float* xr = &xs[row * 128];
    for (int j = 0; j < 128; j++) acc += xr[j] * wqs[j * 4 + h];
    float s = (acc + cvec[h]) * RSQRT128;
    scores[(size_t)n * 4 + h] = s;
    unsigned bits = __float_as_uint(s);
    unsigned key  = (bits & 0x80000000u) ? ~bits : (bits | 0x80000000u);
    atomicMax(&smaxKey[(size_t)zone[n] * 4 + h], key);
  }
}

__global__ void k_smax(const unsigned* __restrict__ key, float* smax, int n4) {
  int t = blockIdx.x * blockDim.x + threadIdx.x;
  if (t >= n4) return;
  unsigned k = key[t];
  float v;
  if (k == 0u) v = 0.f;                                 // empty region (ref: nonfinite -> 0)
  else if (k & 0x80000000u) v = __uint_as_float(k & 0x7fffffffu);
  else v = __uint_as_float(~k);
  smax[t] = v;
}

// ---------------------------------------------------------------------------
// Pass B: per wave, 32-row tile (two 16-row M-subtiles sharing each B-fragment
// load => 2 independent WMMAs per B load, half the L2 B-traffic).
// v_tile = x_tile @ Wv via v_wmma_f32_16x16x32_bf16, then atomic segment-sums.
// ---------------------------------------------------------------------------
__global__ __launch_bounds__(256) void k_attn(
    const float* __restrict__ x, const int* __restrict__ zone,
    const float* __restrict__ scores, const float* __restrict__ smax,
    const __bf16* __restrict__ packedWv, const float* __restrict__ bv,
    float* __restrict__ denom, float* __restrict__ wv, int N) {
  __shared__ float pbuf[8 * 128];
  __shared__ int   zbuf[8 * 32];
  int w = threadIdx.x >> 5, lane = threadIdx.x & 31;
  int rowbase = (blockIdx.x * 8 + w) * 32;
  // phase 1: softmax numerators for this wave's 32 rows x 4 heads
  for (int s = lane; s < 128; s += 32) {
    int rl = s >> 2, h = s & 3;
    int n = rowbase + rl;
    float p = 0.f;
    if (n < N) {
      int z = zone[n];
      float sc = scores[(size_t)n * 4 + h];
      p = __expf(sc - smax[(size_t)z * 4 + h]);
      atomAddF(&denom[(size_t)z * 4 + h], p);
      if (h == 0) zbuf[w * 32 + rl] = z;
    }
    pbuf[w * 128 + s] = p;
  }
  __syncthreads();
  // phase 2: WMMA GEMM v = x_tile @ Wv  (bf16 in, f32 acc), 2 M-subtiles
  int m = lane & 15, kg = lane >> 4;
  int r0 = rowbase + m;      if (r0 >= N) r0 = N - 1;
  int r1 = rowbase + 16 + m; if (r1 >= N) r1 = N - 1;
  const float* x0 = x + (size_t)r0 * 128;
  const float* x1 = x + (size_t)r1 * 128;
  int kbase = kg * 8;
  v8f acc0[8], acc1[8];
  for (int j = 0; j < 8; j++)
    for (int i = 0; i < 8; i++) { acc0[j][i] = 0.f; acc1[j][i] = 0.f; }
#pragma unroll
  for (int kt = 0; kt < 4; kt++) {
    v16bf a0 = make_afrag(x0, kt, kbase);
    v16bf a1 = make_afrag(x1, kt, kbase);
#pragma unroll
    for (int j = 0; j < 8; j++) {
      v16bf b = *(const v16bf*)(packedWv + ((size_t)(kt * 8 + j) * 32 + lane) * 16);
      acc0[j] = __builtin_amdgcn_wmma_f32_16x16x32_bf16(
          false, a0, false, b, (short)0, acc0[j], false, false);
      acc1[j] = __builtin_amdgcn_wmma_f32_16x16x32_bf16(
          false, a1, false, b, (short)0, acc1[j], false, false);
    }
  }
  // phase 3: segment-sum of p * v into wv[R,128] for both subtiles
#pragma unroll
  for (int st = 0; st < 2; st++) {
#pragma unroll
    for (int j = 0; j < 8; j++) {
      int col = j * 16 + m;
      int h = col >> 5;
      float bvc = bv[col];
#pragma unroll
      for (int i = 0; i < 8; i++) {
        int ml = st * 16 + i + kg * 8;
        int n = rowbase + ml;
        if (n < N) {
          float p = pbuf[w * 128 + ml * 4 + h];
          if (p > 0.f) {
            int z = zbuf[w * 32 + ml];
            float v = (st == 0) ? acc0[j][i] : acc1[j][i];
            atomAddF(&wv[(size_t)z * 128 + col], p * (v + bvc));
          }
        }
      }
    }
  }
}

// ---------------------------------------------------------------------------
// O[r] = q + wv[r]/max(denom,1e-9)
// ---------------------------------------------------------------------------
__global__ void k_regionO(const float* __restrict__ qv, const float* __restrict__ wv,
                          const float* __restrict__ denom, float* O, int R) {
  int t = blockIdx.x * blockDim.x + threadIdx.x;
  if (t >= R * 128) return;
  int r = t >> 7, c = t & 127;
  float d = denom[r * 4 + (c >> 5)];
  O[t] = qv[c] + wv[t] / fmaxf(d, 1e-9f);
}

// ---------------------------------------------------------------------------
// Region GEMM (WMMA): mode 0: dst = src + relu(src@W + bias)
//                     mode 1: dst = src@W
// ---------------------------------------------------------------------------
__global__ __launch_bounds__(256) void k_rgemm(
    const float* __restrict__ src, const __bf16* __restrict__ packedW,
    const float* __restrict__ bias, float* __restrict__ dst, int R, int mode) {
  int w = threadIdx.x >> 5, lane = threadIdx.x & 31;
  int tile = blockIdx.x * 8 + w;
  int rowbase = tile * 16;
  if (rowbase >= R) return;
  int m = lane & 15, kg = lane >> 4;
  int rowA = rowbase + m; if (rowA >= R) rowA = R - 1;
  const float* xr = src + (size_t)rowA * 128;
  int kbase = kg * 8;
  v8f acc[8];
  for (int j = 0; j < 8; j++) for (int i = 0; i < 8; i++) acc[j][i] = 0.f;
#pragma unroll
  for (int kt = 0; kt < 4; kt++) {
    v16bf a = make_afrag(xr, kt, kbase);
#pragma unroll
    for (int j = 0; j < 8; j++) {
      v16bf b = *(const v16bf*)(packedW + ((size_t)(kt * 8 + j) * 32 + lane) * 16);
      acc[j] = __builtin_amdgcn_wmma_f32_16x16x32_bf16(
          false, a, false, b, (short)0, acc[j], false, false);
    }
  }
#pragma unroll
  for (int j = 0; j < 8; j++) {
    int col = j * 16 + m;
    float bc = (mode == 0) ? bias[col] : 0.f;
#pragma unroll
    for (int i = 0; i < 8; i++) {
      int row = rowbase + i + kg * 8;
      if (row < R) {
        float v = acc[j][i];
        if (mode == 0) v = src[(size_t)row * 128 + col] + fmaxf(v + bc, 0.f);
        dst[(size_t)row * 128 + col] = v;
      }
    }
  }
}

// --------------------------- GCN pieces ------------------------------------
__global__ void k_deg(const int* __restrict__ ei, unsigned* degCnt, int E) {
  int t = blockIdx.x * blockDim.x + threadIdx.x;
  if (t < E) atomicAdd(&degCnt[ei[E + t]], 1u);
}

__global__ void k_dinv(const unsigned* __restrict__ degCnt, float* dinv, int R) {
  int t = blockIdx.x * blockDim.x + threadIdx.x;
  if (t < R) dinv[t] = rsqrtf((float)(degCnt[t] + 1u));   // +1 for self-loop
}

__global__ void k_edge(const int* __restrict__ ei, const float* __restrict__ dinv,
                       const float* __restrict__ h, float* agg, int E) {
  int t = blockIdx.x * blockDim.x + threadIdx.x;
  if (t >= E * 128) return;
  int e = t >> 7, c = t & 127;
  int r0 = ei[e], c0 = ei[E + e];
  atomAddF(&agg[(size_t)c0 * 128 + c], dinv[r0] * dinv[c0] * h[(size_t)r0 * 128 + c]);
}

__global__ void k_final(const float* __restrict__ agg, const float* __restrict__ h,
                        const float* __restrict__ dinv, const float* __restrict__ bg,
                        const float* __restrict__ pw, float* out, int R) {
  int t = blockIdx.x * blockDim.x + threadIdx.x;
  if (t >= R * 128) return;
  int r = t >> 7, c = t & 127;
  float dv = dinv[r];
  float v = agg[t] + dv * dv * h[t] + bg[c];   // self-loop folded in here
  out[t] = (v >= 0.f) ? v : pw[c] * v;
}

// ---------------------------------------------------------------------------
extern "C" void kernel_launch(void* const* d_in, const int* in_sizes, int n_in,
                              void* d_out, int out_size, void* d_ws, size_t ws_size,
                              hipStream_t stream) {
  const float* x    = (const float*)d_in[0];
  const int*   zone = (const int*)d_in[1];
  const int*   ei   = (const int*)d_in[2];
  const float* S    = (const float*)d_in[4];
  const float* Wq   = (const float*)d_in[5];
  const float* bq   = (const float*)d_in[6];
  const float* Wk   = (const float*)d_in[7];
  const float* bk   = (const float*)d_in[8];
  const float* Wv   = (const float*)d_in[9];
  const float* bv   = (const float*)d_in[10];
  const float* Wo   = (const float*)d_in[11];
  const float* bo   = (const float*)d_in[12];
  const float* Wg   = (const float*)d_in[13];
  const float* bg   = (const float*)d_in[14];
  const float* pw   = (const float*)d_in[15];

  int N = in_sizes[0] / 128;
  int E = in_sizes[2] / 2;
  int R = out_size / 128;

  size_t cur = 0;
  auto take = [&](size_t bytes) -> size_t {
    size_t o = cur;
    cur += (bytes + 255) & ~(size_t)255;
    return o;
  };
  char* ws = (char*)d_ws;
  float*    qv      = (float*)(ws + take(128 * 4));
  float*    wqv     = (float*)(ws + take(512 * 4));
  float*    cvec    = (float*)(ws + take(4 * 4));
  __bf16*   packed  = (__bf16*)(ws + take((size_t)3 * 4 * 8 * 32 * 16 * 2));
  float*    scores  = (float*)(ws + take((size_t)N * 4 * 4));
  unsigned* smaxKey = (unsigned*)(ws + take((size_t)R * 4 * 4));
  float*    smax    = (float*)(ws + take((size_t)R * 4 * 4));
  float*    denom   = (float*)(ws + take((size_t)R * 4 * 4));
  float*    wvb     = (float*)(ws + take((size_t)R * 128 * 4));
  float*    Obuf    = (float*)(ws + take((size_t)R * 128 * 4));
  float*    O2buf   = (float*)(ws + take((size_t)R * 128 * 4));
  float*    hbuf    = (float*)(ws + take((size_t)R * 128 * 4));
  unsigned* degCnt  = (unsigned*)(ws + take((size_t)R * 4));
  float*    dinvb   = (float*)(ws + take((size_t)R * 4));
  float*    aggb    = (float*)(ws + take((size_t)R * 128 * 4));

  // zero accumulators each call (deterministic under graph replay)
  hipMemsetAsync(smaxKey, 0, (size_t)R * 4 * 4, stream);
  hipMemsetAsync(denom,   0, (size_t)R * 4 * 4, stream);
  hipMemsetAsync(wvb,     0, (size_t)R * 128 * 4, stream);
  hipMemsetAsync(degCnt,  0, (size_t)R * 4, stream);
  hipMemsetAsync(aggb,    0, (size_t)R * 128 * 4, stream);

  k_prep<<<1, 128, 0, stream>>>(S, Wq, bq, Wk, bk, qv, wqv, cvec);
  k_pack<<<12, 256, 0, stream>>>(Wv, Wo, Wg, packed);

  k_scores<<<(N + 63) / 64, 256, 0, stream>>>(x, zone, wqv, cvec, scores, smaxKey, N);
  k_smax<<<(R * 4 + 255) / 256, 256, 0, stream>>>(smaxKey, smax, R * 4);

  const __bf16* packWv = packed;
  const __bf16* packWo = packed + (size_t)1 * 4 * 8 * 32 * 16;
  const __bf16* packWg = packed + (size_t)2 * 4 * 8 * 32 * 16;

  k_attn<<<(N + 255) / 256, 256, 0, stream>>>(x, zone, scores, smax, packWv, bv,
                                              denom, wvb, N);
  k_regionO<<<(R * 128 + 255) / 256, 256, 0, stream>>>(qv, wvb, denom, Obuf, R);

  int ntiles = (R + 15) / 16;
  k_rgemm<<<(ntiles + 7) / 8, 256, 0, stream>>>(Obuf, packWo, bo, O2buf, R, 0);
  k_rgemm<<<(ntiles + 7) / 8, 256, 0, stream>>>(O2buf, packWg, bg, hbuf, R, 1);

  k_deg<<<(E + 255) / 256, 256, 0, stream>>>(ei, degCnt, E);
  k_dinv<<<(R + 255) / 256, 256, 0, stream>>>(degCnt, dinvb, R);
  k_edge<<<(E * 128 + 255) / 256, 256, 0, stream>>>(ei, dinvb, hbuf, aggb, E);
  k_final<<<(R * 128 + 255) / 256, 256, 0, stream>>>(aggb, hbuf, dinvb, bg, pw,
                                                     (float*)d_out, R);
}
